// SelfAttention_69784628625808
// MI455X (gfx1250) — matrix operand order repaired and checked by reading, not compiled, not run on hardware
//
#include <hip/hip_runtime.h>
#include <hip/hip_bf16.h>

typedef __attribute__((ext_vector_type(16))) _Float16 v16h;
typedef __attribute__((ext_vector_type(8)))  float    v8f;

union V16H {
    v16h v;
    unsigned u[8];
    _Float16 h[16];
};

#define BATCH 4
#define SEQ   4096
#define DIM   256

// ---------------------------------------------------------------------------
// Kernel 1: QKV projection.  y = x @ W^T  (W is [d_out, d_in], nn.Linear style)
// grid = (SEQ/16, BATCH, 3)  block = 32 (one wave)
// which = 0 -> Q (scaled by 1/sqrt(D), row-major f16)
// which = 1 -> K (row-major f16)
// which = 2 -> V (stored TRANSPOSED: Vt[b][d][s] f16)
// ---------------------------------------------------------------------------
__global__ __launch_bounds__(32)
void qkv_proj_kernel(const float* __restrict__ x,
                     const float* __restrict__ Wq,
                     const float* __restrict__ Wk,
                     const float* __restrict__ Wv,
                     _Float16* __restrict__ Qh,
                     _Float16* __restrict__ Kh,
                     _Float16* __restrict__ Vt)
{
    const int b     = blockIdx.y;
    const int s0    = blockIdx.x * 16;
    const int which = blockIdx.z;                 // uniform across wave
    const float* W  = (which == 0) ? Wq : (which == 1) ? Wk : Wv;

    const int lane = threadIdx.x;                 // 0..31
    const int lm   = lane & 15;
    const int hi   = lane >> 4;

    v8f acc[16];
    const v8f zero = {0.f,0.f,0.f,0.f,0.f,0.f,0.f,0.f};
#pragma unroll
    for (int t = 0; t < 16; ++t) acc[t] = zero;

    const float* xrow = x + ((size_t)b * SEQ + s0 + lm) * DIM;

    for (int kk = 0; kk < 8; ++kk) {              // 8 k-steps of 32
        const int kbase = kk * 32;
        V16H A;
#pragma unroll
        for (int v = 0; v < 8; ++v) {
            const int ko = kbase + 2*v + ((v >= 4) ? 8 : 0) + 8*hi;
            A.h[2*v]   = (_Float16)xrow[ko];
            A.h[2*v+1] = (_Float16)xrow[ko + 1];
        }
#pragma unroll
        for (int t = 0; t < 16; ++t) {
            V16H B;
            const float* wrow = W + (size_t)(t * 16 + lm) * DIM;
#pragma unroll
            for (int v = 0; v < 8; ++v) {
                const int ko = kbase + 2*v + ((v >= 4) ? 8 : 0) + 8*hi;
                B.h[2*v]   = (_Float16)wrow[ko];
                B.h[2*v+1] = (_Float16)wrow[ko + 1];
            }
            acc[t] = __builtin_amdgcn_wmma_f32_16x16x32_f16(
                false, A.v, false, B.v, (short)0, acc[t], false, false);
        }
    }

    if (which == 0) {
        _Float16* dst = Qh + ((size_t)b * SEQ + s0) * DIM;
        const float qscale = 0.0625f;             // 1/sqrt(256)
#pragma unroll
        for (int t = 0; t < 16; ++t)
#pragma unroll
            for (int r = 0; r < 8; ++r) {
                const int m = r + 8*hi;
                dst[(size_t)m * DIM + t*16 + lm] = (_Float16)(acc[t][r] * qscale);
            }
    } else if (which == 1) {
        _Float16* dst = Kh + ((size_t)b * SEQ + s0) * DIM;
#pragma unroll
        for (int t = 0; t < 16; ++t)
#pragma unroll
            for (int r = 0; r < 8; ++r) {
                const int m = r + 8*hi;
                dst[(size_t)m * DIM + t*16 + lm] = (_Float16)acc[t][r];
            }
    } else {
        _Float16* dst = Vt + (size_t)b * DIM * SEQ;
#pragma unroll
        for (int t = 0; t < 16; ++t)
#pragma unroll
            for (int r = 0; r < 8; ++r) {
                const int m = r + 8*hi;
                const int n = t*16 + lm;
                dst[(size_t)n * SEQ + s0 + m] = (_Float16)acc[t][r];
            }
    }
}

// ---------------------------------------------------------------------------
// LDS helpers for async copies
// ---------------------------------------------------------------------------
typedef __attribute__((address_space(3))) const void lds_cvoid;

static __device__ __forceinline__ unsigned lds_addr(const void* p) {
    // ptrtoint of an LDS (addrspace 3) pointer == wave-relative LDS byte offset
    return (unsigned)(unsigned long long)(lds_cvoid*)p;
}

// copy one 64-byte slice (this lane) global -> LDS via the async engine
static __device__ __forceinline__ void async_cp64(unsigned lds, const void* gaddr) {
    asm volatile(
        "global_load_async_to_lds_b128 %0, %1, off\n\t"
        "global_load_async_to_lds_b128 %0, %1, off offset:16\n\t"
        "global_load_async_to_lds_b128 %0, %1, off offset:32\n\t"
        "global_load_async_to_lds_b128 %0, %1, off offset:48"
        :: "v"(lds), "v"(gaddr) : "memory");
}

static __device__ __forceinline__ void wait_async0() {
    asm volatile("s_wait_asynccnt 0x0" ::: "memory");
}

// ---------------------------------------------------------------------------
// Kernel 2: flash attention.
// block = 256 threads (8 waves); each wave owns 16 query rows x full D.
// K / V^T tiles (32 keys) are DOUBLE-BUFFERED in LDS and filled with
// global_load_async_to_lds_b128 so the next tile's fill overlaps this
// tile's 32 WMMAs.  grid = (SEQ/128, BATCH)
// ---------------------------------------------------------------------------
#define KELEM (32 * DIM)              // 8192 f16 per K buffer
#define VELEM (DIM * 32)              // 8192 f16 per V buffer

__global__ __launch_bounds__(256)
void flash_attn_kernel(const _Float16* __restrict__ Qh,
                       const _Float16* __restrict__ Kh,
                       const _Float16* __restrict__ Vt,
                       float* __restrict__ out)
{
    const int b    = blockIdx.y;
    const int tid  = threadIdx.x;
    const int wave = tid >> 5;                    // 0..7
    const int lane = tid & 31;
    const int lm   = lane & 15;
    const int hi   = lane >> 4;
    const int q0   = (blockIdx.x * 8 + wave) * 16;

    // [ K0 | K1 | V0 | V1 | 8 x pbuf ]  = 16+16+16+16+8 KB = 72 KB
    __shared__ __align__(16) _Float16 smem[2*KELEM + 2*VELEM + 8*512];
    _Float16* Kls  = smem;                        // 2 x 8192
    _Float16* Vls  = smem + 2*KELEM;              // 2 x 8192
    _Float16* mypb = smem + 2*KELEM + 2*VELEM + wave*512;

    // Q tile in A layout, resident in registers: 8 k-steps x 8 VGPRs (f16 pairs)
    V16H Qa[8];
    const unsigned* qrow =
        (const unsigned*)(Qh + ((size_t)b * SEQ + q0 + lm) * DIM);
#pragma unroll
    for (int kk = 0; kk < 8; ++kk)
#pragma unroll
        for (int v = 0; v < 8; ++v) {
            const int ko = kk*32 + 2*v + ((v >= 4) ? 8 : 0) + 8*hi;   // even
            Qa[kk].u[v] = qrow[ko >> 1];
        }

    v8f O[16];
    const v8f zero = {0.f,0.f,0.f,0.f,0.f,0.f,0.f,0.f};
#pragma unroll
    for (int t = 0; t < 16; ++t) O[t] = zero;

    float mrow[8], lrow[8];
#pragma unroll
    for (int r = 0; r < 8; ++r) { mrow[r] = -1e30f; lrow[r] = 0.f; }

    // per-thread slice of the cooperative tile fills (64 B of K, 64 B of V)
    const int krow_c = tid >> 3;                  // 0..31 (key row)
    const int kcol_c = (tid & 7) * 32;            // f16 column
    const _Float16* Kbase = Kh + (size_t)b * SEQ * DIM;
    const _Float16* Vbase = Vt + (size_t)b * DIM * SEQ;
    const unsigned kls_slice = lds_addr(Kls + krow_c * DIM + kcol_c);
    const unsigned vls_slice = lds_addr(Vls + tid * 32);

    // prologue: fill buffer 0 with the first key tile
    async_cp64(kls_slice, Kbase + (size_t)krow_c * DIM + kcol_c);
    async_cp64(vls_slice, Vbase + (size_t)tid * SEQ);

    for (int key = 0; key < SEQ; key += 32) {
        const int p  = (key >> 5) & 1;
        const int np = p ^ 1;

        wait_async0();                 // my fills of buffer p have landed
        __syncthreads();               // everyone's fills landed; everyone is
                                       // done reading buffer p^1 (prev body)

        if (key + 32 < SEQ) {          // overlap: fill buffer p^1 w/ next tile
            async_cp64(kls_slice + np * (KELEM * 2),
                       Kbase + (size_t)(key + 32 + krow_c) * DIM + kcol_c);
            async_cp64(vls_slice + np * (VELEM * 2),
                       Vbase + (size_t)tid * SEQ + key + 32);
        }

        // ---- S = Q K^T : two 16x16 tiles (keys key..key+15, key+16..key+31)
        v8f S[2];
#pragma unroll
        for (int j = 0; j < 2; ++j) {
            S[j] = zero;
            const unsigned* krow =
                (const unsigned*)(Kls + p*KELEM + (j*16 + lm) * DIM);
#pragma unroll
            for (int kk = 0; kk < 8; ++kk) {
                V16H Bk;
#pragma unroll
                for (int v = 0; v < 8; ++v) {
                    const int ko = kk*32 + 2*v + ((v >= 4) ? 8 : 0) + 8*hi;   // even
                    Bk.u[v] = krow[ko >> 1];
                }
                S[j] = __builtin_amdgcn_wmma_f32_16x16x32_f16(
                    false, Qa[kk].v, false, Bk.v, (short)0, S[j], false, false);
            }
        }

        // ---- online softmax (row m = r + 8*hi on 16 lanes of a half-wave)
        float scale[8];
#pragma unroll
        for (int r = 0; r < 8; ++r) {
            float mx = fmaxf(S[0][r], S[1][r]);
#pragma unroll
            for (int off = 1; off < 16; off <<= 1)
                mx = fmaxf(mx, __shfl_xor(mx, off, 16));
            const float mnew = fmaxf(mrow[r], mx);
            scale[r] = __expf(mrow[r] - mnew);
            const float p0 = __expf(S[0][r] - mnew);
            const float p1 = __expf(S[1][r] - mnew);
            S[0][r] = p0;
            S[1][r] = p1;
            float rs = p0 + p1;
#pragma unroll
            for (int off = 1; off < 16; off <<= 1)
                rs += __shfl_xor(rs, off, 16);
            lrow[r] = lrow[r] * scale[r] + rs;
            mrow[r] = mnew;
        }

#pragma unroll
        for (int t = 0; t < 16; ++t)
#pragma unroll
            for (int r = 0; r < 8; ++r)
                O[t][r] *= scale[r];

        // ---- convert P from C layout (f32) to A layout (f16) via LDS
#pragma unroll
        for (int r = 0; r < 8; ++r) {
            const int m = r + 8*hi;
            mypb[m*32 + lm]      = (_Float16)S[0][r];
            mypb[m*32 + 16 + lm] = (_Float16)S[1][r];
        }
        __syncthreads();               // order pbuf store -> load

        V16H Pa;
        const unsigned* pb = (const unsigned*)mypb;
#pragma unroll
        for (int v = 0; v < 8; ++v) {
            const int ko = 2*v + ((v >= 4) ? 8 : 0) + 8*hi;           // even
            Pa.u[v] = pb[(lm*32 + ko) >> 1];
        }

        // ---- O += P @ V : B[k][n] = V[key+k][16t+n] = Vt[16t+n][key+k]
#pragma unroll
        for (int t = 0; t < 16; ++t) {
            V16H Bv;
            const unsigned* vrow =
                (const unsigned*)(Vls + p*VELEM + (t*16 + lm) * 32);
#pragma unroll
            for (int v = 0; v < 8; ++v) {
                const int ko = 2*v + ((v >= 4) ? 8 : 0) + 8*hi;           // even
                Bv.u[v] = vrow[ko >> 1];
            }
            O[t] = __builtin_amdgcn_wmma_f32_16x16x32_f16(
                false, Pa.v, false, Bv.v, (short)0, O[t], false, false);
        }
    }

    // ---- finalize: divide by softmax denominator and store f32
    float* dst = out + ((size_t)b * SEQ + q0) * DIM;
#pragma unroll
    for (int r = 0; r < 8; ++r) {
        const float inv = 1.0f / lrow[r];
        const int m = r + 8*hi;
#pragma unroll
        for (int t = 0; t < 16; ++t)
            dst[(size_t)m * DIM + t*16 + lm] = O[t][r] * inv;
    }
}

// ---------------------------------------------------------------------------
extern "C" void kernel_launch(void* const* d_in, const int* in_sizes, int n_in,
                              void* d_out, int out_size, void* d_ws, size_t ws_size,
                              hipStream_t stream)
{
    const float* x  = (const float*)d_in[0];
    const float* Wq = (const float*)d_in[1];
    const float* Wk = (const float*)d_in[2];
    const float* Wv = (const float*)d_in[3];
    float* out = (float*)d_out;

    const size_t elems = (size_t)BATCH * SEQ * DIM;   // 4 Mi elements
    _Float16* Qh = (_Float16*)d_ws;                    // 8 MB
    _Float16* Kh = Qh + elems;                         // 8 MB
    _Float16* Vt = Kh + elems;                         // 8 MB (transposed V)

    dim3 g1(SEQ / 16, BATCH, 3);
    qkv_proj_kernel<<<g1, 32, 0, stream>>>(x, Wq, Wk, Wv, Qh, Kh, Vt);

    dim3 g2(SEQ / 128, BATCH, 1);
    flash_attn_kernel<<<g2, 256, 0, stream>>>(Qh, Kh, Vt, out);
}
